// Screen2Vec_12515534701228
// MI455X (gfx1250) — matrix operand order repaired and checked by reading, compiled, not run on hardware
//
#include <hip/hip_runtime.h>
#include <hip/hip_bf16.h>
#include <stdint.h>

// Problem constants (from reference): B=16, S=64, T=64, D=768
#define D_ 768
#define S_ 64
#define T_ 64
#define BM 128      // rows per block tile
#define BN 64       // cols per block tile
#define BK 64       // K chunk == 2 WMMA k-steps per stage
#define LDA 72      // padded LDS stride (bf16 elems); 144B per row, 16B-aligned
#define LDB 72      // matches TDM pad: 128B row + 16B pad = 144B pitch

typedef __attribute__((ext_vector_type(16))) __bf16 v16bf;
typedef __attribute__((ext_vector_type(8)))  float  v8f;
typedef __attribute__((ext_vector_type(4)))  unsigned v4u;
typedef __attribute__((ext_vector_type(8)))  unsigned v8u;

union Frag { v16bf v; uint4 q[2]; };

// Pack two fp32 -> packed bf16 pair (round-half-up) with one v_perm_b32.
__device__ __forceinline__ unsigned pk2(float a, float b) {
  unsigned ra = __float_as_uint(a) + 0x8000u;
  unsigned rb = __float_as_uint(b) + 0x8000u;
  return __builtin_amdgcn_perm(rb, ra, 0x07060302);
}

// gfx1250 async copy: 16B global -> LDS, tracked by ASYNCcnt.
__device__ __forceinline__ void async_b128_to_lds(void* lds_ptr, const void* gptr) {
  unsigned lds_off = (unsigned)(uintptr_t)lds_ptr;   // low 32 bits of generic = LDS offset
  asm volatile("global_load_async_to_lds_b128 %0, %1, off"
               :: "v"(lds_off), "v"(gptr) : "memory");
}
__device__ __forceinline__ void wait_asynccnt0() {
  asm volatile("s_wait_asynccnt 0x0" ::: "memory");
}

// gfx1250 Tensor Data Mover: load a (tile_n x tile_k) bf16 2D tile (row pitch
// row_stride_elems in global) into LDS at lds_off, with 16B pad per 128B row
// so LDS pitch == 144B (LDB=72 elems). ISA 08 / D# groups 0-1; VADDR2/3 NULL.
__device__ __forceinline__ void tdm_load_tile_b(
    const unsigned short* gsrc, unsigned lds_off,
    unsigned row_stride_elems, unsigned tile_k, unsigned tile_n)
{
  unsigned long long ga = (unsigned long long)gsrc;
  v4u g0;
  g0[0] = 1u;                                   // count=1, is_restore=0, no gather
  g0[1] = lds_off;                              // lds_addr (bytes)
  g0[2] = (unsigned)ga;                         // global_addr[31:0]
  g0[3] = (unsigned)(ga >> 32) | (2u << 30);    // global_addr[56:32] | type=2
  v8u g1;
  g1[0] = (1u << 16)                            // data_size = 1 -> 2 bytes
        | (1u << 20)                            // pad_enable
        | (4u << 22)                            // pad_interval: 32 DWORDs (128B)
        | (3u << 25);                           // pad_amount: 4 DWORDs (16B)
  g1[1] = (row_stride_elems & 0xFFFFu) << 16;   // tensor_dim0[15:0]   (bits 79:48)
  g1[2] = (row_stride_elems >> 16)              // tensor_dim0[31:16]
        | (0x4000u << 16);                      // tensor_dim1[15:0] = 16384 (ample)
  g1[3] = (tile_k << 16);                       // tensor_dim1[31:16]=0 | tile_dim0
  g1[4] = tile_n;                               // tile_dim1 (tile_dim2 = 0)
  g1[5] = row_stride_elems;                     // tensor_dim0_stride[31:0]
  g1[6] = 0;                                    // stride[47:32], dim1_stride lo
  g1[7] = 0;
  asm volatile("tensor_load_to_lds %0, %1" :: "s"(g0), "s"(g1) : "memory");
}

// A fragment: 16x32 bf16 tile at k-offset ks, rows r0..r0+15 (ISA 7.12.2 A layout).
__device__ __forceinline__ v16bf load_a_frag(const unsigned short* As, int r0, int ks, int lane) {
  Frag f;
  const int m = lane & 15, half = lane >> 4;
  const unsigned short* base = As + (r0 + m) * LDA + ks;
  f.q[0] = *(const uint4*)(base + half * 8);
  f.q[1] = *(const uint4*)(base + 16 + half * 8);
  return f.v;
}
// B fragment: 32x16 bf16 ([n][k] in LDS); lanes 0-15 K=0..7,16..23, lanes 16-31 K=8..15,24..31.
__device__ __forceinline__ v16bf load_b_frag(const unsigned short* Bs, int n0, int ks, int lane) {
  Frag f;
  const int n = lane & 15, half = lane >> 4;
  const unsigned short* base = Bs + (n0 + n) * LDB + ks;
  f.q[0] = *(const uint4*)(base + half * 8);
  f.q[1] = *(const uint4*)(base + 16 + half * 8);
  return f.v;
}

__global__ void cvt_bf16_kernel(const float* __restrict__ src,
                                unsigned short* __restrict__ dst, int n) {
  int i = blockIdx.x * blockDim.x + threadIdx.x;
  if (i < n) dst[i] = (unsigned short)((__float_as_uint(src[i]) + 0x8000u) >> 16);
}

// Shared GEMM stage + compute for one BK-chunk.
// A source fp32 (converted on the fly); B source bf16 [n][stride] staged either
// by the Tensor Data Mover (TENSORcnt) or per-lane async copies (ASYNCcnt).
template <bool USE_TDM>
__device__ __forceinline__ void gemm_chunk(
    unsigned short* As, unsigned short* Bs,
    const float* a0src, const float* a1src,  // two row-region base ptrs
    int a_rows_lo,                           // rows [0, a_rows_lo) use a0src, rest a1src
    long a_stride,
    const unsigned short* Bsrc, long b_stride, int n0, int kk,
    int tid, int lane, int wr, int wc, v8f acc[2][2], bool prefetch_next)
{
  const int wave = tid >> 5;
  // ---- B tile: 64 rows x 64 bf16 ----
  if (USE_TDM) {
    if (wave == 0) {
      tdm_load_tile_b(Bsrc + (long)n0 * b_stride + kk,
                      (unsigned)(uintptr_t)Bs, (unsigned)b_stride, BK, BN);
    }
  } else {
    #pragma unroll
    for (int it = 0; it < 2; ++it) {
      int idx  = tid + it * 256;
      int nrow = idx >> 3;
      int ck   = (idx & 7) << 3;
      async_b128_to_lds(&Bs[nrow * LDB + ck],
                        Bsrc + (long)(n0 + nrow) * b_stride + kk + ck);
    }
  }
  // ---- A tile: 128 rows x 64 fp32 -> bf16; batch loads, then convert+store ----
  float4 fr[8];
  #pragma unroll
  for (int it = 0; it < 8; ++it) {
    int idx = tid + it * 256;
    int row = idx >> 4;                 // 16 float4 chunks per row
    int ck  = (idx & 15) << 2;
    const float* base = (row < a_rows_lo) ? a0src : a1src;
    int rrow = (row < a_rows_lo) ? row : (row - a_rows_lo);
    const float* src = base + (long)rrow * a_stride + kk + ck;
    fr[it] = *(const float4*)src;
    if (prefetch_next) __builtin_prefetch(src + BK, 0, 1);   // global_prefetch_b8
  }
  #pragma unroll
  for (int it = 0; it < 8; ++it) {
    int idx = tid + it * 256;
    int row = idx >> 4;
    int ck  = (idx & 15) << 2;
    unsigned* dst = (unsigned*)&As[row * LDA + ck];
    dst[0] = pk2(fr[it].x, fr[it].y);
    dst[1] = pk2(fr[it].z, fr[it].w);
  }
  if (USE_TDM) {
    if (wave == 0) __builtin_amdgcn_s_wait_tensorcnt(0);
  } else {
    wait_asynccnt0();
  }
  __syncthreads();
  // ---- compute: 2 k-steps x 4 WMMA ----
  #pragma unroll
  for (int ks = 0; ks < BK; ks += 32) {
    v16bf a0 = load_a_frag(As, wr,      ks, lane);
    v16bf a1 = load_a_frag(As, wr + 16, ks, lane);
    v16bf b0 = load_b_frag(Bs, wc,      ks, lane);
    v16bf b1 = load_b_frag(Bs, wc + 16, ks, lane);
    acc[0][0] = __builtin_amdgcn_wmma_f32_16x16x32_bf16(false, a0, false, b0,
                  (short)0, acc[0][0], false, false);
    acc[0][1] = __builtin_amdgcn_wmma_f32_16x16x32_bf16(false, a0, false, b1,
                  (short)0, acc[0][1], false, false);
    acc[1][0] = __builtin_amdgcn_wmma_f32_16x16x32_bf16(false, a1, false, b0,
                  (short)0, acc[1][0], false, false);
    acc[1][1] = __builtin_amdgcn_wmma_f32_16x16x32_bf16(false, a1, false, b1,
                  (short)0, acc[1][1], false, false);
  }
  __syncthreads();
}

// One RNN step, fused:  h_out = mask ? tanh(UIs_s @ Wih^T + h_in @ Whh^T + b) : h_in
__global__ __launch_bounds__(256)
void rnn_step(const float* __restrict__ UIs, const float* __restrict__ h_in,
              const unsigned short* __restrict__ Wih,
              const unsigned short* __restrict__ Whh,
              const float* __restrict__ b_ih, const float* __restrict__ b_hh,
              const int* __restrict__ tsl, float* __restrict__ h_out, int s)
{
  __shared__ __align__(16) unsigned short As[BM * LDA];
  __shared__ __align__(16) unsigned short Bs[BN * LDB];
  const int tid  = threadIdx.x;
  const int lane = tid & 31, wave = tid >> 5;
  const int wr = (wave & 3) * 32, wc = (wave >> 2) * 32;
  const int m0 = blockIdx.x * BM, n0 = blockIdx.y * BN;

  v8f acc[2][2] = {};

  // phase 0: A rows are UIs[b, s, t, :] with row = b*T + t.
  // A 128-row block tile spans exactly two b values (T=64): split at row 64.
  {
    const float* u0 = UIs + (long)((m0 >> 6) + 0) * (S_ * T_ * D_) + (long)s * (T_ * D_);
    const float* u1 = UIs + (long)((m0 >> 6) + 1) * (S_ * T_ * D_) + (long)s * (T_ * D_);
    for (int kk = 0; kk < D_; kk += BK)
      gemm_chunk<true>(As, Bs, u0, u1, 64, D_, Wih, D_, n0, kk,
                       tid, lane, wr, wc, acc, kk + BK < D_);
  }
  // phase 1: A rows from h_in (contiguous)
  {
    const float* h = h_in + (long)m0 * D_;
    for (int kk = 0; kk < D_; kk += BK)
      gemm_chunk<true>(As, Bs, h, h, BM, D_, Whh, D_, n0, kk,
                       tid, lane, wr, wc, acc, kk + BK < D_);
  }

  // ---- epilogue: bias + tanh + length mask ----
  const int nn = lane & 15, half = lane >> 4;
  #pragma unroll
  for (int i = 0; i < 2; ++i) {
    int rbase = m0 + wr + i * 16 + half * 8;
    #pragma unroll
    for (int j = 0; j < 2; ++j) {
      int col = n0 + wc + j * 16 + nn;
      float bias = b_ih[col] + b_hh[col];
      #pragma unroll
      for (int v = 0; v < 8; ++v) {
        int row  = rbase + v;
        float hn = tanhf(acc[i][j][v] + bias);
        bool valid = s < tsl[row];                       // tsl is (B,T) == row-flat
        h_out[(long)row * D_ + col] = valid ? hn : h_in[(long)row * D_ + col];
      }
    }
  }
}

// out = concat(h_final, descr) @ Wlin^T + b_lin   (M=1024, N=768, K=1536)
__global__ __launch_bounds__(256)
void final_linear(const float* __restrict__ h_fin, const float* __restrict__ descr,
                  const unsigned short* __restrict__ Wlin,
                  const float* __restrict__ b_lin, float* __restrict__ out)
{
  __shared__ __align__(16) unsigned short As[BM * LDA];
  __shared__ __align__(16) unsigned short Bs[BN * LDB];
  const int tid  = threadIdx.x;
  const int lane = tid & 31, wave = tid >> 5;
  const int wr = (wave & 3) * 32, wc = (wave >> 2) * 32;
  const int m0 = blockIdx.x * BM, n0 = blockIdx.y * BN;

  v8f acc[2][2] = {};

  #pragma unroll 1
  for (int phase = 0; phase < 2; ++phase) {
    const float* Asrc = (phase ? descr : h_fin) + (long)m0 * D_;
    for (int kk = 0; kk < D_; kk += BK)
      gemm_chunk<false>(As, Bs, Asrc, Asrc, BM, D_, Wlin + (long)phase * D_, 2 * D_,
                        n0, kk, tid, lane, wr, wc, acc, kk + BK < D_);
  }

  const int nn = lane & 15, half = lane >> 4;
  #pragma unroll
  for (int i = 0; i < 2; ++i) {
    int rbase = m0 + wr + i * 16 + half * 8;
    #pragma unroll
    for (int j = 0; j < 2; ++j) {
      int col = n0 + wc + j * 16 + nn;
      float bias = b_lin[col];
      #pragma unroll
      for (int v = 0; v < 8; ++v) {
        int row = rbase + v;
        out[(long)row * D_ + col] = acc[i][j][v] + bias;
      }
    }
  }
}

extern "C" void kernel_launch(void* const* d_in, const int* in_sizes, int n_in,
                              void* d_out, int out_size, void* d_ws, size_t ws_size,
                              hipStream_t stream) {
  (void)in_sizes; (void)n_in; (void)out_size; (void)ws_size;
  const float* UIs   = (const float*)d_in[0];
  const float* descr = (const float*)d_in[1];
  const int*   tsl   = (const int*)  d_in[2];
  const float* Wih   = (const float*)d_in[3];
  const float* Whh   = (const float*)d_in[4];
  const float* bih   = (const float*)d_in[5];
  const float* bhh   = (const float*)d_in[6];
  const float* Wlin  = (const float*)d_in[7];
  const float* blin  = (const float*)d_in[8];
  float* out = (float*)d_out;

  // Workspace layout (~10.5 MiB)
  char* ws = (char*)d_ws;
  unsigned short* Wih16  = (unsigned short*)ws; ws += (size_t)768 * 768 * 2;
  unsigned short* Whh16  = (unsigned short*)ws; ws += (size_t)768 * 768 * 2;
  unsigned short* Wlin16 = (unsigned short*)ws; ws += (size_t)768 * 1536 * 2;
  float* h0 = (float*)ws; ws += (size_t)1024 * 768 * 4;
  float* h1 = (float*)ws;

  cvt_bf16_kernel<<<(768 * 768  + 255) / 256, 256, 0, stream>>>(Wih,  Wih16,  768 * 768);
  cvt_bf16_kernel<<<(768 * 768  + 255) / 256, 256, 0, stream>>>(Whh,  Whh16,  768 * 768);
  cvt_bf16_kernel<<<(768 * 1536 + 255) / 256, 256, 0, stream>>>(Wlin, Wlin16, 768 * 1536);
  hipMemsetAsync(h0, 0, (size_t)1024 * 768 * 4, stream);

  float* hin = h0; float* hout = h1;
  for (int s = 0; s < S_; ++s) {
    rnn_step<<<dim3(1024 / BM, D_ / BN), 256, 0, stream>>>(
        UIs, hin, Wih16, Whh16, bih, bhh, tsl, hout, s);
    float* t = hin; hin = hout; hout = t;
  }
  // 64 swaps -> final hidden is back in h0 (== hin)
  final_linear<<<dim3(1024 / BM, D_ / BN), 256, 0, stream>>>(hin, descr, Wlin16, blin, out);
}
